// Attention_5033701671198
// MI455X (gfx1250) — matrix operand order repaired and checked by reading, compiled, not run on hardware
//
#include <hip/hip_runtime.h>
#include <hip/hip_bf16.h>

typedef __bf16 bf16;
typedef __attribute__((ext_vector_type(16))) __bf16 v16bf;
typedef __attribute__((ext_vector_type(8)))  float  v8f;

#define D_LATENT 128
#define N_HEADS  16
#define SEQ      2048
#define BATCH    2
#define INNER    (N_HEADS * D_LATENT)   // 2048
#define MROWS    (BATCH * SEQ)          // 4096

// ---------------------------------------------------------------------------
// WMMA helpers (CDNA5 bf16 16x16x32, f32 accumulate)
// ---------------------------------------------------------------------------
__device__ __forceinline__ v8f wmma_bf16(v16bf a, v16bf b, v8f c) {
    return __builtin_amdgcn_wmma_f32_16x16x32_bf16(
        /*neg_a=*/false, a, /*neg_b=*/false, b,
        /*c_mod=*/(short)0, c, /*reuse_a=*/false, /*reuse_b=*/false);
}

// A-fragment (16x32 bf16, MxK): lane row = lane&15, hi = lane>>4.
// element e -> K = kbase + hi*16 + e (contiguous, matches B fragments)
__device__ __forceinline__ v16bf load_a_frag(const bf16* rowp, int kbase, int hi) {
    v16bf a;
    const bf16* p0 = rowp + kbase + hi * 16;
    const bf16* p1 = rowp + kbase + hi * 16 + 8;
#pragma unroll
    for (int j = 0; j < 8; ++j) { a[j] = p0[j]; a[8 + j] = p1[j]; }
    return a;
}

// B-fragment (32x16 bf16, KxN): lane col = lane&15, khalf = lane>>4,
// elements = 16 contiguous K values -> one 32-byte contiguous load.
__device__ __forceinline__ v16bf load_b_frag_contig(const bf16* p) {
    v16bf b;
#pragma unroll
    for (int e = 0; e < 16; ++e) b[e] = p[e];
    return b;
}

__device__ __forceinline__ v8f zero_v8f() {
    v8f c;
#pragma unroll
    for (int j = 0; j < 8; ++j) c[j] = 0.0f;
    return c;
}

__device__ __forceinline__ float half_reduce_max(float x) {
#pragma unroll
    for (int off = 1; off < 16; off <<= 1) x = fmaxf(x, __shfl_xor(x, off, 32));
    return x;
}
__device__ __forceinline__ float half_reduce_sum(float x) {
#pragma unroll
    for (int off = 1; off < 16; off <<= 1) x += __shfl_xor(x, off, 32);
    return x;
}

// Async global->LDS copy (CDNA5, tracked by ASYNCcnt)
__device__ __forceinline__ void async_copy_b128(unsigned lds_off, const void* gptr) {
    asm volatile("global_load_async_to_lds_b128 %0, %1, off"
                 :: "v"(lds_off), "v"((unsigned long long)gptr)
                 : "memory");
}
template <int N>
__device__ __forceinline__ void wait_async_le() {
    asm volatile("s_wait_asynccnt %0" :: "i"(N) : "memory");
}

// ---------------------------------------------------------------------------
// Kernel 0: one-time weight pre-convert + transpose to bf16, N-major.
//  Wq/Wk/Wv [128x2048] f32 -> WT [2048x128] bf16 ; Wo [2048x128] -> WoT [128x2048]
// ---------------------------------------------------------------------------
__global__ void transpose_weights_kernel(const float* __restrict__ Wq,
                                         const float* __restrict__ Wk,
                                         const float* __restrict__ Wv,
                                         const float* __restrict__ Wo,
                                         bf16* __restrict__ WqT,
                                         bf16* __restrict__ WkT,
                                         bf16* __restrict__ WvT,
                                         bf16* __restrict__ WoT) {
    size_t idx   = (size_t)blockIdx.x * blockDim.x + threadIdx.x;
    size_t which = idx >> 18;            // 262144 elements per matrix
    size_t e     = idx & 262143;
    if (which < 3) {
        const float* W = (which == 0) ? Wq : (which == 1) ? Wk : Wv;
        bf16*        T = (which == 0) ? WqT : (which == 1) ? WkT : WvT;
        int k = (int)(e >> 11);          // 0..127
        int n = (int)(e & 2047);         // 0..2047
        T[(size_t)n * D_LATENT + k] = (bf16)W[e];
    } else {
        int k = (int)(e >> 7);           // 0..2047
        int n = (int)(e & 127);          // 0..127
        WoT[(size_t)n * INNER + k] = (bf16)Wo[e];
    }
}

// ---------------------------------------------------------------------------
// Kernel 1: fused QKV projection + RoPE.  X[4096x128] @ W[128x2048] (x3)
// Writes Qh/Kh (RoPE'd) [B,H,S,D] and VhT [B,H,D,S] (d-major for PV fragments).
// ---------------------------------------------------------------------------
__global__ void qkv_rope_kernel(const float* __restrict__ q,
                                const bf16* __restrict__ WqT,
                                const bf16* __restrict__ WkT,
                                const bf16* __restrict__ WvT,
                                bf16* __restrict__ Qh,
                                bf16* __restrict__ Kh,
                                bf16* __restrict__ VhT) {
    __shared__ bf16 As[16][128];   // 4 KB

    const int bx  = blockIdx.x;
    const int w   = bx / (256 * 16);        // 0=Q, 1=K, 2=V
    const int rem = bx % (256 * 16);
    const int mt  = rem >> 4;               // m-tile 0..255
    const int ng  = rem & 15;               // 128-col group 0..15
    const int m0  = mt * 16;
    const bf16* WT = (w == 0) ? WqT : (w == 1) ? WkT : WvT;

    for (int i = threadIdx.x; i < 16 * 128; i += blockDim.x) {
        int r = i >> 7, k = i & 127;
        As[r][k] = (bf16)q[(m0 + r) * D_LATENT + k];
    }
    __syncthreads();

    const int wave = threadIdx.x >> 5;
    const int lane = threadIdx.x & 31;
    const int hi   = lane >> 4;
    const int ln   = lane & 15;
    const int col  = ng * 128 + wave * 16 + ln;   // output column

    v8f c = zero_v8f();
    const bf16* wcol = WT + (size_t)col * D_LATENT;   // K-contiguous column
#pragma unroll
    for (int kc = 0; kc < 4; ++kc) {        // K = 128 -> 4 steps of 32
        v16bf a = load_a_frag(&As[ln][0], kc * 32, hi);
        v16bf b = load_b_frag_contig(wcol + kc * 32 + hi * 16);
        c = wmma_bf16(a, b, c);
    }

    const int h = col >> 7;                 // head
    const int d = col & 127;                // dim within head
    if (w == 2) {                           // V: no RoPE, store d-major [bh][d][s]
#pragma unroll
        for (int r = 0; r < 8; ++r) {
            int m = m0 + r + hi * 8;
            int b_ = m >> 11, s = m & (SEQ - 1);
            VhT[(((size_t)b_ * N_HEADS + h) * D_LATENT + d) * SEQ + s] = (bf16)c[r];
        }
    } else {                                // Q/K: interleaved-pair RoPE
        bf16* Out = (w == 0) ? Qh : Kh;
        const int   p    = d >> 1;
        const float invf = __expf(-(float)(2 * p) * (9.210340371976184f / 128.0f));
#pragma unroll
        for (int r = 0; r < 8; ++r) {
            int m = m0 + r + hi * 8;
            int b_ = m >> 11, s = m & (SEQ - 1);
            float ang = (float)s * invf;
            float cs = __cosf(ang), sn = __sinf(ang);
            float x     = c[r];
            float other = __shfl_xor(x, 1, 32);       // pair partner (d even<->odd)
            float res = (d & 1) ? (x * cs + other * sn)    // r1 = x1*c + x0*s
                                : (x * cs - other * sn);   // r0 = x0*c - x1*s
            Out[(((size_t)b_ * N_HEADS + h) * SEQ + s) * D_LATENT + d] = (bf16)res;
        }
    }
}

// ---------------------------------------------------------------------------
// Kernel 2: causal flash attention, one wave per (b,h, 16-row q-tile).
// 64-key blocks, mask-free main loop + masked diagonal tail.
// ---------------------------------------------------------------------------
#define ATT_WAVES 8

template <bool MASKED>
__device__ __forceinline__ void attn_block64(const v16bf aq[4], v8f acc[8],
                                             float mrow[8], float lrow[8],
                                             const bf16* __restrict__ Kb,
                                             const bf16* __restrict__ VbT,
                                             bf16 (*PstW)[64],
                                             int key0, int q0, int hi, int ln) {
    const float scale = 0.08838834764831845f;   // 1/sqrt(128)

    // Scores: four 16x16 tiles (keys key0 .. key0+63), 16 WMMAs
    v8f s[4];
#pragma unroll
    for (int t = 0; t < 4; ++t) s[t] = zero_v8f();
#pragma unroll
    for (int kc = 0; kc < 4; ++kc) {
#pragma unroll
        for (int t = 0; t < 4; ++t) {
            v16bf b = load_b_frag_contig(
                Kb + (size_t)(key0 + t * 16 + ln) * D_LATENT + kc * 32 + hi * 16);
            s[t] = wmma_bf16(aq[kc], b, s[t]);
        }
    }

    // Online softmax (rows constant across each 16-lane half)
#pragma unroll
    for (int r = 0; r < 8; ++r) {
        const int R = q0 + r + hi * 8;                // global query row
        float v[4];
#pragma unroll
        for (int t = 0; t < 4; ++t) {
            v[t] = s[t][r] * scale;
            if (MASKED && (key0 + t * 16 + ln > R)) v[t] = -__builtin_inff();
        }
        float mx = fmaxf(fmaxf(v[0], v[1]), fmaxf(v[2], v[3]));
        mx = half_reduce_max(mx);
        float mnew = fmaxf(mrow[r], mx);
        float p[4], sum = 0.0f;
#pragma unroll
        for (int t = 0; t < 4; ++t) { p[t] = __expf(v[t] - mnew); sum += p[t]; }
        sum = half_reduce_sum(sum);
        float alpha = __expf(mrow[r] - mnew);
        mrow[r] = mnew;
        lrow[r] = lrow[r] * alpha + sum;
#pragma unroll
        for (int dc = 0; dc < 8; ++dc) acc[dc][r] *= alpha;
#pragma unroll
        for (int t = 0; t < 4; ++t) PstW[r + hi * 8][t * 16 + ln] = (bf16)p[t];
    }
    asm volatile("s_wait_dscnt 0" ::: "memory");      // same-wave store->load order

    // P (16x64) as two A-fragments; V fragments contiguous from d-major VhT
    v16bf ap0 = load_a_frag(&PstW[ln][0], 0,  hi);
    v16bf ap1 = load_a_frag(&PstW[ln][0], 32, hi);
#pragma unroll
    for (int dc = 0; dc < 8; ++dc) {
        const bf16* vrow = VbT + (size_t)(dc * 16 + ln) * SEQ + key0 + hi * 16;
        v16bf bv0 = load_b_frag_contig(vrow);
        v16bf bv1 = load_b_frag_contig(vrow + 32);
        acc[dc] = wmma_bf16(ap0, bv0, acc[dc]);
        acc[dc] = wmma_bf16(ap1, bv1, acc[dc]);
    }
}

__global__ void flash_attn_kernel(const bf16* __restrict__ Qh,
                                  const bf16* __restrict__ Kh,
                                  const bf16* __restrict__ VhT,
                                  bf16* __restrict__ AttnOut) {
    __shared__ bf16 Pst[ATT_WAVES][16][64];   // 16 KB total

    const int wave = threadIdx.x >> 5;
    const int lane = threadIdx.x & 31;
    const int hi   = lane >> 4;
    const int ln   = lane & 15;

    const int task = blockIdx.x * ATT_WAVES + wave;  // 0..4095
    const int bh   = task >> 7;                      // b*H + h
    const int q0   = (task & 127) * 16;              // first query row of tile

    const bf16* Qb  = Qh  + (size_t)bh * SEQ * D_LATENT;
    const bf16* Kb  = Kh  + (size_t)bh * SEQ * D_LATENT;
    const bf16* VbT = VhT + (size_t)bh * D_LATENT * SEQ;

    // Q tile as 4 A-fragments (registers), row = lane&15
    v16bf aq[4];
    {
        const bf16* qrow = Qb + (size_t)(q0 + ln) * D_LATENT;
#pragma unroll
        for (int kc = 0; kc < 4; ++kc) aq[kc] = load_a_frag(qrow, kc * 32, hi);
    }

    v8f acc[8];
#pragma unroll
    for (int dc = 0; dc < 8; ++dc) acc[dc] = zero_v8f();
    float mrow[8], lrow[8];
#pragma unroll
    for (int r = 0; r < 8; ++r) { mrow[r] = -__builtin_inff(); lrow[r] = 0.0f; }

    // Mask-free full blocks strictly below the diagonal tile
    const int nfull = q0 >> 6;
    for (int kb = 0; kb < nfull; ++kb)
        attn_block64<false>(aq, acc, mrow, lrow, Kb, VbT, &Pst[wave][0],
                            kb * 64, q0, hi, ln);
    // Masked tail covering keys up to the diagonal (exactly 1 block)
    for (int key0 = nfull * 64; key0 <= q0 + 15; key0 += 64)
        attn_block64<true>(aq, acc, mrow, lrow, Kb, VbT, &Pst[wave][0],
                           key0, q0, hi, ln);

    // Epilogue: one reciprocal per row, then scale + store bf16 AttnOut [B,S,H*D]
    float linv[8];
#pragma unroll
    for (int r = 0; r < 8; ++r) linv[r] = __builtin_amdgcn_rcpf(lrow[r]);
    const int b_ = bh >> 4, h = bh & 15;
#pragma unroll
    for (int dc = 0; dc < 8; ++dc) {
#pragma unroll
        for (int r = 0; r < 8; ++r) {
            const int R = q0 + r + hi * 8;
            AttnOut[((size_t)b_ * SEQ + R) * INNER + h * D_LATENT + dc * 16 + ln] =
                (bf16)(acc[dc][r] * linv[r]);
        }
    }
}

// ---------------------------------------------------------------------------
// Kernel 3: output projection  AttnOut[4096x2048](bf16) @ W_o[2048x128] -> f32
// Double-buffered async global->LDS staging (ASYNCcnt pipeline).
// ---------------------------------------------------------------------------
__device__ __forceinline__ void stage_chunk_async(bf16 (*buf)[256],
                                                  const bf16* __restrict__ src,
                                                  int m0, int kcb, int tid) {
    for (int i = tid; i < 16 * 32; i += 256) {       // 512 B128 transfers, 2/thread
        int r = i >> 5, cg = i & 31;
        unsigned loff = (unsigned)(unsigned long long)&buf[r][cg * 8];
        async_copy_b128(loff, src + (size_t)(m0 + r) * INNER + kcb * 256 + cg * 8);
    }
}

__global__ void out_proj_kernel(const bf16* __restrict__ AttnOut,
                                const bf16* __restrict__ WoT,
                                float* __restrict__ out) {
    __shared__ bf16 Ast[2][16][256];   // 2 x 8 KB double buffer

    const int m0   = blockIdx.x * 16;
    const int wave = threadIdx.x >> 5;
    const int lane = threadIdx.x & 31;
    const int hi   = lane >> 4;
    const int ln   = lane & 15;
    const int n0   = wave * 16;
    const bf16* wcol = WoT + (size_t)(n0 + ln) * INNER;   // K-contiguous column

    v8f c = zero_v8f();
    stage_chunk_async(Ast[0], AttnOut, m0, 0, threadIdx.x);   // prime chunk 0

    for (int kcb = 0; kcb < 8; ++kcb) {          // K = 2048 in chunks of 256
        if (kcb + 1 < 8) {
            stage_chunk_async(Ast[(kcb + 1) & 1], AttnOut, m0, kcb + 1, threadIdx.x);
            wait_async_le<2>();                  // chunk kcb done (in-order), kcb+1 in flight
        } else {
            wait_async_le<0>();
        }
        __syncthreads();
#pragma unroll
        for (int kc = 0; kc < 8; ++kc) {
            v16bf a = load_a_frag(&Ast[kcb & 1][ln][0], kc * 32, hi);
            v16bf b = load_b_frag_contig(wcol + kcb * 256 + kc * 32 + hi * 16);
            c = wmma_bf16(a, b, c);
        }
        __syncthreads();
    }
#pragma unroll
    for (int r = 0; r < 8; ++r)
        out[(size_t)(m0 + r + hi * 8) * D_LATENT + n0 + ln] = c[r];
}

// ---------------------------------------------------------------------------
extern "C" void kernel_launch(void* const* d_in, const int* in_sizes, int n_in,
                              void* d_out, int out_size, void* d_ws, size_t ws_size,
                              hipStream_t stream) {
    (void)in_sizes; (void)n_in; (void)out_size; (void)ws_size;
    const float* q  = (const float*)d_in[0];
    const float* Wq = (const float*)d_in[1];
    const float* Wk = (const float*)d_in[2];
    const float* Wv = (const float*)d_in[3];
    const float* Wo = (const float*)d_in[4];

    const size_t per  = (size_t)BATCH * N_HEADS * SEQ * D_LATENT;  // 8M bf16
    const size_t wper = (size_t)INNER * D_LATENT;                  // 256K bf16
    bf16* Qh      = (bf16*)d_ws;
    bf16* Kh      = Qh + per;
    bf16* VhT     = Kh + per;          // [B,H,D,S]
    bf16* AttnOut = VhT + per;         // [B,S,INNER]
    bf16* WqT     = AttnOut + per;     // [INNER, D_LATENT]
    bf16* WkT     = WqT + wper;
    bf16* WvT     = WkT + wper;
    bf16* WoT     = WvT + wper;        // [D_LATENT, INNER]

    transpose_weights_kernel<<<(4 * 262144) / 256, 256, 0, stream>>>(
        Wq, Wk, Wv, Wo, WqT, WkT, WvT, WoT);
    qkv_rope_kernel<<<256 * 16 * 3, 256, 0, stream>>>(q, WqT, WkT, WvT, Qh, Kh, VhT);
    flash_attn_kernel<<<4096 / ATT_WAVES, ATT_WAVES * 32, 0, stream>>>(Qh, Kh, VhT, AttnOut);
    out_proj_kernel<<<MROWS / 16, 256, 0, stream>>>(AttnOut, WoT, (float*)d_out);
}
